// GDN_28226525070378
// MI455X (gfx1250) — matrix operand order repaired and checked by reading, not compiled, not run on hardware
//
#include <hip/hip_runtime.h>
#include <hip/hip_bf16.h>
#include <stdint.h>

// GDN: y[b,c,hw] = x / sqrt(|beta|+eps + sum_j |gamma[c,j]| * x[b,j,hw]^2)
// B=8, C=192, HW=65536 -> 524288 pixels. Per-block: 128 pixels x 192 chans.

typedef __bf16  v16bf __attribute__((ext_vector_type(16)));
typedef __bf16  v8bf  __attribute__((ext_vector_type(8)));
typedef float   v8f   __attribute__((ext_vector_type(8)));

#define C_CH    192
#define G_STR   200          // bf16 elems per padded gamma row (400B, 16B aligned)
#define X_STR   132          // f32 elems per padded x row (528B, 16B aligned)
#define TILE_P  128          // pixels per block
#define LDS_G_BYTES   (192 * G_STR * 2)            // 76800
#define LDS_X_BYTES   (192 * X_STR * 4)            // 101376
#define LDS_B_BYTES   (192 * 4)                    // 768
#define LDS_TOTAL     (LDS_G_BYTES + LDS_X_BYTES + LDS_B_BYTES)  // 178944

static __device__ __forceinline__ __bf16 f2bf(float f) {
    union { float f; uint32_t u; } v; v.f = f;
    uint32_t r = v.u + 0x7FFFu + ((v.u >> 16) & 1u);   // round-to-nearest-even
    uint16_t h = (uint16_t)(r >> 16);
    __bf16 o;
    __builtin_memcpy(&o, &h, 2);
    return o;
}

extern "C" __global__ __launch_bounds__(256)
void gdn_wmma_kernel(const float* __restrict__ x,
                     const float* __restrict__ beta,
                     const float* __restrict__ gamma,
                     float* __restrict__ out) {
    extern __shared__ char smem[];
    __bf16* Gl = (__bf16*)smem;                          // [192][G_STR] bf16 |gamma|
    float*  XT = (float*)(smem + LDS_G_BYTES);           // [192][X_STR] f32 x tile
    float*  BB = (float*)(smem + LDS_G_BYTES + LDS_X_BYTES); // [192] |beta|+eps

    const int tid  = threadIdx.x;
    const int w    = tid >> 5;     // wave 0..7
    const int lane = tid & 31;

    const int p0  = blockIdx.x * TILE_P;     // global pixel base
    const int b   = p0 >> 16;                // batch index (tiles never straddle)
    const int hw0 = p0 & 65535;

    // ---------- stage |gamma| -> LDS bf16 (L2-resident source) ----------
    for (int r = w; r < C_CH; r += 8) {
        const float* gsrc = gamma + r * C_CH;
        __bf16* gdst = Gl + r * G_STR;
        #pragma unroll
        for (int q = 0; q < 6; ++q) {
            int j = lane + 32 * q;
            gdst[j] = f2bf(__builtin_fabsf(gsrc[j]));
        }
    }
    if (tid < C_CH) BB[tid] = __builtin_fabsf(beta[tid]) + 1e-6f;

    // ---------- stage x tile via async global->LDS DMA (ASYNCcnt) ----------
    {
        const int c0 = w * 24;  // 24 channel rows per wave
        const float* src0 = x + (((size_t)(b * C_CH + c0)) << 16) + hw0 + lane * 4;
        unsigned ldsoff0 = (unsigned)(uintptr_t)(XT + c0 * X_STR + lane * 4);
        #pragma unroll 4
        for (int i = 0; i < 24; ++i) {
            const float* s = src0 + ((size_t)i << 16);
            unsigned d = ldsoff0 + (unsigned)(i * X_STR * 4);
            asm volatile("global_load_async_to_lds_b128 %0, %1, off"
                         :: "v"(d), "v"(s) : "memory");
        }
        asm volatile("s_wait_asynccnt 0" ::: "memory");
    }
    __syncthreads();

    // ---------- per-wave 16-pixel strip: norm = G @ (x^2) via WMMA ----------
    const int n     = lane & 15;      // pixel column within strip / A row
    const int hi    = lane >> 4;      // half-wave
    const int pb    = w * 16;         // strip pixel base in tile
    const int khalf = hi * 16;        // unused; B-fragment K offset is kq0 (line 92)
    const int kq0   = hi * 8;         // A-fragment K offset for upper half

    // Build the six K=32 B-fragments: s = bf16(x^2), layout:
    // lanes0-15 K=[0..7],[16..23]; lanes16-31 +8, two bf16 per VGPR.
    v16bf bfrag[6];
    #pragma unroll
    for (int kc = 0; kc < 6; ++kc) {
        const int kk = kc * 32;
        v16bf bf;
        #pragma unroll
        for (int e = 0; e < 16; ++e) {
            float v = XT[(kk + kq0 + e + ((e >> 3) << 3)) * X_STR + pb + n];
            bf[e] = f2bf(v * v);
        }
        bfrag[kc] = bf;
    }

    // 12 M-tiles of 16 output channels each.
    for (int m = 0; m < 12; ++m) {
        v8f acc = {0.f, 0.f, 0.f, 0.f, 0.f, 0.f, 0.f, 0.f};
        const __bf16* grow = Gl + (m * 16 + n) * G_STR;   // A row M = lane%16
        #pragma unroll
        for (int kc = 0; kc < 6; ++kc) {
            const int kk = kc * 32;
            // A layout: lanes0-15 K=[kk..kk+7],[kk+16..kk+23]; lanes16-31 +8.
            v8bf alo = *(const v8bf*)(grow + kk + kq0);
            v8bf ahi = *(const v8bf*)(grow + kk + 16 + kq0);
            v16bf af;
            #pragma unroll
            for (int e = 0; e < 8; ++e) { af[e] = alo[e]; af[e + 8] = ahi[e]; }
            acc = __builtin_amdgcn_wmma_f32_16x16x32_bf16(
                      false, af, false, bfrag[kc], (short)0, acc, false, false);
        }
        // Epilogue: C/D layout -> channel c = m*16 + hi*8 + r, pixel = pb+n.
        #pragma unroll
        for (int r = 0; r < 8; ++r) {
            int   c    = m * 16 + hi * 8 + r;
            float norm = acc[r] + BB[c];
            float xv   = XT[c * X_STR + pb + n];
            float y    = xv * __builtin_amdgcn_rsqf(norm);
            out[(((size_t)(b * C_CH + c)) << 16) + (size_t)(hw0 + pb + n)] = y;
        }
    }
}

extern "C" void kernel_launch(void* const* d_in, const int* in_sizes, int n_in,
                              void* d_out, int out_size, void* d_ws, size_t ws_size,
                              hipStream_t stream) {
    (void)n_in; (void)d_ws; (void)ws_size; (void)out_size;
    const float* x     = (const float*)d_in[0];
    const float* beta  = (const float*)d_in[1];
    const float* gamma = (const float*)d_in[2];
    float*       out   = (float*)d_out;

    // 178944 B dynamic LDS per block (WGP has 320KB).
    static int attr_done = 0;
    (void)attr_done;
    hipFuncSetAttribute((const void*)gdn_wmma_kernel,
                        hipFuncAttributeMaxDynamicSharedMemorySize, LDS_TOTAL);

    const int total_pixels = in_sizes[0] / C_CH;          // 524288
    const int nblocks      = total_pixels / TILE_P;       // 4096
    gdn_wmma_kernel<<<nblocks, 256, LDS_TOTAL, stream>>>(x, beta, gamma, out);
}
